// GCN_17411797418393
// MI455X (gfx1250) — matrix operand order repaired and checked by reading, not compiled, not run on hardware
//
#include <hip/hip_runtime.h>
#include <hip/hip_bf16.h>

// ---------------------------------------------------------------------------
// 2-layer GCN (PyG GCNConv style) for MI455X / gfx1250.
//   layer: deg^{-1/2} sym-norm, h = x@W (bf16 WMMA, f32 acc), atomic scatter,
//          +bias (+relu / +log_softmax).
// All intermediate node buffers use a padded stride of 128 so the WMMA GEMM
// runs guard-free with b128 vector loads and immediate-offset stores.
// ---------------------------------------------------------------------------

#define GCN_N 40000
#define GCN_E 640000
#define GCN_D 125
#define GCN_LD 128     // padded feature stride

typedef __attribute__((ext_vector_type(16))) __bf16 v16bf;
typedef __attribute__((ext_vector_type(8)))  float  v8f;

// --------------------------- degree / norm ---------------------------------
__global__ __launch_bounds__(256) void gcn_deg_init(float* deg) {
    int n = blockIdx.x * 256 + threadIdx.x;
    if (n < GCN_N) deg[n] = 1.0f;                 // self loop
}

__global__ __launch_bounds__(256) void gcn_deg_count(const int* __restrict__ dst,
                                                     float* __restrict__ deg) {
    int e = blockIdx.x * 256 + threadIdx.x;
    if (e < GCN_E) unsafeAtomicAdd(&deg[dst[e]], 1.0f);
}

__global__ __launch_bounds__(256) void gcn_deg_rsqrt(float* deg) {
    int n = blockIdx.x * 256 + threadIdx.x;
    if (n < GCN_N) {
        float d = deg[n];
        deg[n] = (d > 0.0f) ? __frsqrt_rn(d) : 0.0f;
    }
}

// --------------------------- pad x -> [N,128] ------------------------------
__global__ __launch_bounds__(256)
void gcn_pad(const float* __restrict__ x, float* __restrict__ xp) {
    size_t i = (size_t)blockIdx.x * 256 + threadIdx.x;
    if (i >= (size_t)GCN_N * GCN_LD) return;
    int n = (int)(i >> 7), c = (int)(i & 127);
    xp[i] = (c < GCN_D) ? x[(size_t)n * GCN_D + c] : 0.0f;
}

// --------------------------- GEMM: H = X @ W (WMMA bf16) -------------------
// X: [N,128] f32 padded, W: [125,125] f32 row-major, H: [N,128] f32.
// Block = 256 threads = 8 waves; each wave computes a 16-row x 128-col tile:
// 8 N-tiles x 4 K-chunks = 32 v_wmma_f32_16x16x32_bf16 per wave.
// W staged in LDS transposed as zero-padded 128x128 bf16, row stride 144
// elements (288B): every B fragment is one 32B-aligned v16bf (2x ds_load_b128),
// conflict-free (288B == full 64-bank row + 8-bank skew).
__global__ __launch_bounds__(256)
void gcn_gemm_wmma(const float* __restrict__ X, const float* __restrict__ W,
                   float* __restrict__ H) {
    __shared__ __align__(32) __bf16 Wlds[128 * 144];

    const int tid = threadIdx.x;
    for (int i = tid; i < 128 * 128; i += 256) {
        int k = i >> 7;         // K row of W
        int n = i & 127;        // output column
        float w = (k < GCN_D && n < GCN_D) ? W[k * GCN_D + n] : 0.0f;
        Wlds[n * 144 + k] = (__bf16)w;          // transposed store
    }
    __syncthreads();

    const int lane = tid & 31;
    const int wave = tid >> 5;
    const int hi   = lane >> 4;       // half-wave select
    const int mrow = lane & 15;
    const int rowTile = blockIdx.x * 8 + wave;
    if (rowTile >= GCN_N / 16) return;   // wave-uniform: EXEC all-1s for WMMA

    // A fragments, contiguous K order (matches B):
    //  elems 0..15 : K = kc*32 + 16*hi + e     (16 contiguous floats)
    // X padded to 128 stride -> unconditional 16B-aligned float4 loads.
    const float4* xr4 =
        (const float4*)(X + (size_t)(rowTile * 16 + mrow) * GCN_LD);

    v16bf afrag[4];
#pragma unroll
    for (int kc = 0; kc < 4; ++kc) {
        float4 p0 = xr4[kc * 8 + 4 * hi];
        float4 p1 = xr4[kc * 8 + 4 * hi + 1];
        float4 p2 = xr4[kc * 8 + 4 * hi + 2];
        float4 p3 = xr4[kc * 8 + 4 * hi + 3];
        afrag[kc][0]  = (__bf16)p0.x; afrag[kc][1]  = (__bf16)p0.y;
        afrag[kc][2]  = (__bf16)p0.z; afrag[kc][3]  = (__bf16)p0.w;
        afrag[kc][4]  = (__bf16)p1.x; afrag[kc][5]  = (__bf16)p1.y;
        afrag[kc][6]  = (__bf16)p1.z; afrag[kc][7]  = (__bf16)p1.w;
        afrag[kc][8]  = (__bf16)p2.x; afrag[kc][9]  = (__bf16)p2.y;
        afrag[kc][10] = (__bf16)p2.z; afrag[kc][11] = (__bf16)p2.w;
        afrag[kc][12] = (__bf16)p3.x; afrag[kc][13] = (__bf16)p3.y;
        afrag[kc][14] = (__bf16)p3.z; afrag[kc][15] = (__bf16)p3.w;
    }

    // C/D layout: VGPR i -> row (i + 8*hi), col = lane&15. One base address,
    // all stores use compile-time immediate offsets.
    float* hp = H + (size_t)rowTile * 16 * GCN_LD + (size_t)hi * 8 * GCN_LD + mrow;

#pragma unroll
    for (int t = 0; t < 8; ++t) {
        const int col = t * 16 + mrow;
        v8f acc = {};
#pragma unroll
        for (int kc = 0; kc < 4; ++kc) {
            // B fragment: lanes 0-15 hold K=kc*32..+15, lanes 16-31 K=+16..+31;
            // contiguous in transposed LDS -> one 32B v16bf load.
            const v16bf bfrag =
                *(const v16bf*)&Wlds[col * 144 + kc * 32 + hi * 16];
            acc = __builtin_amdgcn_wmma_f32_16x16x32_bf16(
                false, afrag[kc], false, bfrag, (short)0, acc, false, false);
        }
#pragma unroll
        for (int i = 0; i < 8; ++i)
            hp[t * 16 + i * GCN_LD] = acc[i];   // pad cols come out exactly 0
    }
}

// --------------------------- aggregation -----------------------------------
// out[n] = h[n] * dinv[n]^2   (self-loop term; also initializes the buffer)
template <int LDO>
__global__ __launch_bounds__(256)
void gcn_self_loop(const float* __restrict__ H, const float* __restrict__ dinv,
                   float* __restrict__ out) {
    size_t i = (size_t)blockIdx.x * 256 + threadIdx.x;
    if (i >= (size_t)GCN_N * LDO) return;
    int n, c;
    if (LDO == GCN_LD) { n = (int)(i >> 7); c = (int)(i & 127); }
    else               { n = (int)(i / LDO); c = (int)(i - (size_t)n * LDO); }
    float dv = dinv[n];
    out[i] = H[(size_t)n * GCN_LD + c] * (dv * dv);
}

// One wave32 per edge; lane handles columns {lane, lane+32, lane+64, lane+96}.
// For LDO==128 the guard vanishes: pad columns of H are 0, adding 0 is free.
template <int LDO, bool GUARD>
__global__ __launch_bounds__(256)
void gcn_edge_scatter(const float* __restrict__ H, const int* __restrict__ src,
                      const int* __restrict__ dst, const float* __restrict__ dinv,
                      float* __restrict__ out) {
    int gw   = (int)(((size_t)blockIdx.x * 256 + threadIdx.x) >> 5);
    int lane = threadIdx.x & 31;
    if (gw >= GCN_E) return;
    int s = src[gw], d = dst[gw];
    float norm = dinv[s] * dinv[d];
    const float* hs = H + (size_t)s * GCN_LD;
    float* od = out + (size_t)d * LDO;
#pragma unroll
    for (int j = 0; j < 4; ++j) {
        int c = lane + j * 32;
        if (!GUARD || c < GCN_D) unsafeAtomicAdd(&od[c], hs[c] * norm);
    }
}

__global__ __launch_bounds__(256)
void gcn_bias_relu(float* __restrict__ a, const float* __restrict__ b) {
    size_t i = (size_t)blockIdx.x * 256 + threadIdx.x;
    if (i >= (size_t)GCN_N * GCN_LD) return;
    int c = (int)(i & 127);
    float v = (c < GCN_D) ? (a[i] + b[c]) : 0.0f;   // keep pad columns at 0
    a[i] = fmaxf(v, 0.0f);
}

// One wave32 per node: out[n] = log_softmax(out[n] + b)   (stride 125, in place)
__global__ __launch_bounds__(256)
void gcn_bias_logsoftmax(float* __restrict__ out, const float* __restrict__ b) {
    int node = (int)(((size_t)blockIdx.x * 256 + threadIdx.x) >> 5);
    int lane = threadIdx.x & 31;
    if (node >= GCN_N) return;
    float* o = out + (size_t)node * GCN_D;

    float v[4];
    float m = -3.0e38f;
#pragma unroll
    for (int j = 0; j < 4; ++j) {
        int c = lane + j * 32;
        v[j] = (c < GCN_D) ? (o[c] + b[c]) : -3.0e38f;
        m = fmaxf(m, v[j]);
    }
#pragma unroll
    for (int off = 16; off > 0; off >>= 1) m = fmaxf(m, __shfl_xor(m, off, 32));

    float s = 0.0f;
#pragma unroll
    for (int j = 0; j < 4; ++j) {
        int c = lane + j * 32;
        if (c < GCN_D) s += __expf(v[j] - m);
    }
#pragma unroll
    for (int off = 16; off > 0; off >>= 1) s += __shfl_xor(s, off, 32);

    float ls = __logf(s);
#pragma unroll
    for (int j = 0; j < 4; ++j) {
        int c = lane + j * 32;
        if (c < GCN_D) o[c] = v[j] - m - ls;
    }
}

// --------------------------- launch ----------------------------------------
extern "C" void kernel_launch(void* const* d_in, const int* in_sizes, int n_in,
                              void* d_out, int out_size, void* d_ws, size_t ws_size,
                              hipStream_t stream) {
    (void)in_sizes; (void)n_in; (void)out_size; (void)ws_size;

    const float* x  = (const float*)d_in[0];        // [N, 125]
    const int*   ei = (const int*)d_in[1];          // [2, E] (int32 on device)
    const float* W1 = (const float*)d_in[2];        // [125, 125]
    const float* b1 = (const float*)d_in[3];        // [125]
    const float* W2 = (const float*)d_in[4];        // [125, 125]
    const float* b2 = (const float*)d_in[5];        // [125]
    const int* src = ei;
    const int* dst = ei + GCN_E;

    float* out  = (float*)d_out;                            // [N, 125]
    // ws: xpad/a1 alias (xpad dead after GEMM-1; kernels are stream-ordered)
    float* xpad = (float*)d_ws;                             // [N, 128]
    float* a1   = xpad;                                     // [N, 128]
    float* h    = xpad + (size_t)GCN_N * GCN_LD;            // [N, 128]
    float* dinv = h    + (size_t)GCN_N * GCN_LD;            // [N]

    const int nodeBlocks = (GCN_N + 255) / 256;
    const int edgeBlocks = (GCN_E + 255) / 256;
    const int e128Blocks = (int)(((size_t)GCN_N * GCN_LD + 255) / 256);
    const int e125Blocks = (int)(((size_t)GCN_N * GCN_D + 255) / 256);
    const int waveEdgeBlocks = (int)(((size_t)GCN_E * 32 + 255) / 256);
    const int waveNodeBlocks = (int)(((size_t)GCN_N * 32 + 255) / 256);
    const int gemmBlocks = (GCN_N / 16 + 7) / 8;     // 8 waves/block, 16 rows/wave

    // Symmetric normalization: dinv = rsqrt(1 + indeg)
    gcn_deg_init  <<<nodeBlocks, 256, 0, stream>>>(dinv);
    gcn_deg_count <<<edgeBlocks, 256, 0, stream>>>(dst, dinv);
    gcn_deg_rsqrt <<<nodeBlocks, 256, 0, stream>>>(dinv);

    // ---- layer 1 ----
    gcn_pad       <<<e128Blocks, 256, 0, stream>>>(x, xpad);
    gcn_gemm_wmma <<<gemmBlocks, 256, 0, stream>>>(xpad, W1, h);
    gcn_self_loop<GCN_LD>          <<<e128Blocks, 256, 0, stream>>>(h, dinv, a1);
    gcn_edge_scatter<GCN_LD, false><<<waveEdgeBlocks, 256, 0, stream>>>(h, src, dst, dinv, a1);
    gcn_bias_relu <<<e128Blocks, 256, 0, stream>>>(a1, b1);

    // ---- layer 2 (accumulate straight into d_out, softmax in place) ----
    gcn_gemm_wmma <<<gemmBlocks, 256, 0, stream>>>(a1, W2, h);
    gcn_self_loop<GCN_D>          <<<e125Blocks, 256, 0, stream>>>(h, dinv, out);
    gcn_edge_scatter<GCN_D, true> <<<waveEdgeBlocks, 256, 0, stream>>>(h, src, dst, dinv, out);
    gcn_bias_logsoftmax<<<waveNodeBlocks, 256, 0, stream>>>(out, b2);
}
